// SimplifiedEdgeConv_21208548508406
// MI455X (gfx1250) — matrix operand order repaired and checked by reading, not compiled, not run on hardware
//
#include <hip/hip_runtime.h>
#include <hip/hip_bf16.h>

#define BB 8
#define CC 64
#define NN 4096
#define KN 20
#define OO 64
#define TWO_C 128

typedef __attribute__((ext_vector_type(16))) __bf16 v16bf;
typedef __attribute__((ext_vector_type(8)))  float  v8f;

union BFrag { v16bf v; uint4 q[2]; };

// ---------------------------------------------------------------------------
// Kernel 1: transpose x[B,C,N] -> xt[B,N,C] (bf16) and squared norms xx[B,N]
// ---------------------------------------------------------------------------
__global__ void k_transpose(const float* __restrict__ x,
                            __hip_bfloat16* __restrict__ xt,
                            float* __restrict__ xx) {
    int t = blockIdx.x * blockDim.x + threadIdx.x;   // 0 .. B*N-1
    int b = t >> 12;
    int n = t & (NN - 1);
    const float* xp = x + (size_t)b * CC * NN + n;
    __hip_bfloat16* row = xt + (size_t)t * CC;
    float s = 0.f;
#pragma unroll
    for (int c = 0; c < CC; ++c) {
        float v = xp[(size_t)c * NN];                // coalesced across threads
        s += v * v;
        row[c] = __float2bfloat16(v);
    }
    xx[t] = s;
}

// ---------------------------------------------------------------------------
// Kernel 2: convert W to bf16 (row-contiguous in K) and zero the BN stat bins
// ---------------------------------------------------------------------------
__global__ void k_prep(const float* __restrict__ W,
                       __hip_bfloat16* __restrict__ Wt,
                       float* __restrict__ stats) {
    int t = threadIdx.x;
    for (int i = t; i < OO * TWO_C; i += 256) Wt[i] = __float2bfloat16(W[i]);
    if (t < 128) stats[t] = 0.f;                     // sum[64] + sumsq[64]
}

// ---------------------------------------------------------------------------
// Kernel 3: fused Gram-matrix (WMMA bf16) + streaming per-row top-20 kNN.
// One wave handles a 32-row tile (two 16x16 WMMA D tiles sharing B frags);
// every lane owns one row's top-20 list. The row-constant ||xi||^2 term is
// dropped (ordering-invariant) and the column term ||xj||^2 is folded into
// the score before the LDS transpose, so the selection loop is pure compares.
// ---------------------------------------------------------------------------
__global__ __launch_bounds__(32) void k_knn(const __hip_bfloat16* __restrict__ xt,
                                            const float* __restrict__ xx,
                                            int* __restrict__ idx) {
    __shared__ float tile[32 * 16];            // 32 rows x 16 cols of scores

    const int b  = blockIdx.x >> 7;            // 128 row tiles (of 32) per batch
    const int n0 = (blockIdx.x & 127) << 5;
    const int l  = threadIdx.x;
    const int lm = l & 15;
    const int hi = l >> 4;                     // 0 or 1
    const __hip_bfloat16* xb = xt + (size_t)b * NN * CC;

    // A fragments for two row tiles (rows n0..n0+15 and n0+16..n0+31).
    // ISA 16-bit A layout: lane<16: M=l, K halfs {32s+0..7},{32s+16..23};
    // lane>=16: M=l-16, chunks shifted by +8 halfs.
    BFrag a0[2], a1[2];
    {
        const uint4* r0 = reinterpret_cast<const uint4*>(xb + (size_t)(n0 + lm) * CC);
        const uint4* r1 = reinterpret_cast<const uint4*>(xb + (size_t)(n0 + 16 + lm) * CC);
#pragma unroll
        for (int s = 0; s < 2; ++s) {
            a0[s].q[0] = r0[s * 4 + hi];      a0[s].q[1] = r0[s * 4 + 2 + hi];
            a1[s].q[0] = r1[s * 4 + hi];      a1[s].q[1] = r1[s * 4 + 2 + hi];
        }
    }

    float bd[KN]; int bi[KN];                  // lane l owns row n0+l
#pragma unroll
    for (int j = 0; j < KN; ++j) { bd[j] = 3.4e38f; bi[j] = 0; }

    for (int mt = 0; mt < NN / 16; ++mt) {
        const int m0 = mt << 4;
        // B fragments (shared by both row tiles): lane = column m0+lm,
        // K halfs {32s+8*hi..+7},{32s+16+8*hi..+7} (matches A layout).
        BFrag bf[2];
        const uint4* colp = reinterpret_cast<const uint4*>(xb + (size_t)(m0 + lm) * CC);
#pragma unroll
        for (int s = 0; s < 2; ++s) {
            bf[s].q[0] = colp[s * 4 + hi];
            bf[s].q[1] = colp[s * 4 + 2 + hi];
        }
        if (mt + 1 < NN / 16)                  // global_prefetch_b8
            __builtin_prefetch(xb + (size_t)(m0 + 16 + lm) * CC, 0, 0);
        const float xxc = xx[b * NN + m0 + lm];

        v8f c0 = {}, c1 = {};
        c0 = __builtin_amdgcn_wmma_f32_16x16x32_bf16(false, a0[0].v, false, bf[0].v,
                                                     (short)0, c0, false, false);
        c0 = __builtin_amdgcn_wmma_f32_16x16x32_bf16(false, a0[1].v, false, bf[1].v,
                                                     (short)0, c0, false, false);
        c1 = __builtin_amdgcn_wmma_f32_16x16x32_bf16(false, a1[0].v, false, bf[0].v,
                                                     (short)0, c1, false, false);
        c1 = __builtin_amdgcn_wmma_f32_16x16x32_bf16(false, a1[1].v, false, bf[1].v,
                                                     (short)0, c1, false, false);

        // Fold column norm into the score, transpose via LDS.
        // D layout: lane = column lm, VGPR v = row v + 8*hi.
#pragma unroll
        for (int v = 0; v < 8; ++v) {
            tile[(v + 8 * hi) * 16 + lm]        = fmaf(-2.0f, c0[v], xxc);
            tile[(16 + v + 8 * hi) * 16 + lm]   = fmaf(-2.0f, c1[v], xxc);
        }
        __syncthreads();

#pragma unroll
        for (int cc2 = 0; cc2 < 16; ++cc2) {   // all 32 lanes select
            float d = tile[l * 16 + cc2];
            int   id = m0 + cc2;
            if (d < bd[KN - 1]) {              // sorted-list insertion
#pragma unroll
                for (int j = 0; j < KN; ++j) {
                    bool  sm = d < bd[j];
                    float td = bd[j]; int ti = bi[j];
                    bd[j] = sm ? d : td;  bi[j] = sm ? id : ti;
                    d     = sm ? td : d;  id    = sm ? ti : id;
                }
            }
        }
        __syncthreads();
    }
    {
        int* op = idx + (size_t)(b * NN + n0 + l) * KN;
#pragma unroll
        for (int j = 0; j < KN; ++j) op[j] = bi[j];
    }
}

// ---------------------------------------------------------------------------
// Kernels 4/6: edge-conv GEMM via WMMA bf16. PASS==1 accumulates BN stats
// (sum, sumsq per channel); PASS==2 normalizes + LeakyReLU + max over k.
// Block = 1 wave, handles 4 points (80 edge rows = 5 WMMA row tiles).
// Weights are staged to LDS with GLOBAL_LOAD_ASYNC_TO_LDS_B128 (ASYNCcnt).
// ---------------------------------------------------------------------------
template <int PASS>
__global__ __launch_bounds__(32) void k_conv(const __hip_bfloat16* __restrict__ xt,
                                             const int* __restrict__ idx,
                                             const __hip_bfloat16* __restrict__ Wt,
                                             float* __restrict__ stats,
                                             float* __restrict__ out) {
    __shared__ __hip_bfloat16 wlds[OO * TWO_C];   // 16 KB: staged weights
    __shared__ __hip_bfloat16 elds[16 * TWO_C];   //  4 KB: one edge row tile
    __shared__ float hbuf[80 * OO];               // 20 KB: activations (pass 2)

    const int l  = threadIdx.x;
    const int lm = l & 15, hi = l >> 4;
    const int b  = blockIdx.x >> 10;              // 1024 blocks per batch
    const int n0 = (blockIdx.x & 1023) << 2;      // 4 points per block

    {   // async DMA of W into LDS, bypassing VGPRs (tracked by ASYNCcnt).
        // Generic shared-aperture addresses carry the LDS byte offset in the
        // low 32 bits (ISA 10.2 aperture mapping).
        uint32_t lbase = (uint32_t)(uintptr_t)(&wlds[0]);
        const char* gbase = (const char*)Wt;
        for (int i = l; i < OO * TWO_C * 2 / 16; i += 32) {
            uint32_t    laddr = lbase + i * 16;
            const char* g     = gbase + (size_t)i * 16;
            asm volatile("global_load_async_to_lds_b128 %0, %1, off"
                         :: "v"(laddr), "v"(g) : "memory");
        }
        asm volatile("s_wait_asynccnt 0" ::: "memory");
    }
    float scale[4], shift[4];
    if (PASS == 2) {
#pragma unroll
        for (int ot = 0; ot < 4; ++ot) {
            scale[ot] = stats[128 + ot * 16 + lm];
            shift[ot] = stats[192 + ot * 16 + lm];
        }
    }
    float asum[4] = {0, 0, 0, 0}, asq[4] = {0, 0, 0, 0};
    __syncthreads();

    const __hip_bfloat16* xb = xt + (size_t)b * NN * CC;
    for (int t = 0; t < 5; ++t) {
        {   // build edge tile: 2 lanes per row, 32 channels (16 bf16 pairs)
            int rr = l >> 1, p = l & 1;
            int gr = t * 16 + rr;
            int nl = gr / KN, kk = gr % KN;
            int n  = n0 + nl;
            int nbr = idx[(size_t)(b * NN + n) * KN + kk];
            const __hip_bfloat162* cp =
                reinterpret_cast<const __hip_bfloat162*>(xb + (size_t)n   * CC + p * 32);
            const __hip_bfloat162* gp =
                reinterpret_cast<const __hip_bfloat162*>(xb + (size_t)nbr * CC + p * 32);
            __hip_bfloat162* ed =
                reinterpret_cast<__hip_bfloat162*>(elds + rr * TWO_C + p * 32);
            __hip_bfloat162* ec =
                reinterpret_cast<__hip_bfloat162*>(elds + rr * TWO_C + 64 + p * 32);
#pragma unroll
            for (int j = 0; j < 16; ++j) {
                __hip_bfloat162 cv = cp[j];
                __hip_bfloat162 gv = gp[j];
                __hip_bfloat162 df;
                df.x = __float2bfloat16(__bfloat162float(gv.x) - __bfloat162float(cv.x));
                df.y = __float2bfloat16(__bfloat162float(gv.y) - __bfloat162float(cv.y));
                ed[j] = df;
                ec[j] = cv;
            }
        }
        __syncthreads();

        // A fragments for this tile: 4 K=32 steps over the 128-wide edge row
        BFrag a[4];
        {
            const uint4* rp = reinterpret_cast<const uint4*>(elds + lm * TWO_C);
#pragma unroll
            for (int s = 0; s < 4; ++s) {
                a[s].q[0] = rp[s * 4 + hi];
                a[s].q[1] = rp[s * 4 + 2 + hi];
            }
        }
#pragma unroll
        for (int ot = 0; ot < 4; ++ot) {          // 4 output column tiles
            v8f c = {};
#pragma unroll
            for (int s = 0; s < 4; ++s) {
                BFrag bfr;
                const uint4* wp = reinterpret_cast<const uint4*>(
                    wlds + (ot * 16 + lm) * TWO_C + s * 32 + hi * 8);
                bfr.q[0] = wp[0];
                bfr.q[1] = wp[2];
                c = __builtin_amdgcn_wmma_f32_16x16x32_bf16(false, a[s].v, false, bfr.v,
                                                            (short)0, c, false, false);
            }
            if (PASS == 1) {
#pragma unroll
                for (int v = 0; v < 8; ++v) { asum[ot] += c[v]; asq[ot] += c[v] * c[v]; }
            } else {
#pragma unroll
                for (int v = 0; v < 8; ++v) {
                    int   row = t * 16 + v + 8 * hi;
                    float hn  = c[v] * scale[ot] + shift[ot];
                    float act = hn >= 0.f ? hn : 0.2f * hn;   // LeakyReLU(0.2)
                    hbuf[row * OO + ot * 16 + lm] = act;
                }
            }
        }
        __syncthreads();
    }

    if (PASS == 1) {
#pragma unroll
        for (int ot = 0; ot < 4; ++ot) {   // fold rows 8..15 (lanes 16..31)
            asum[ot] += __shfl_xor(asum[ot], 16, 32);
            asq[ot]  += __shfl_xor(asq[ot], 16, 32);
        }
        if (hi == 0) {
#pragma unroll
            for (int ot = 0; ot < 4; ++ot) {
                atomicAdd(&stats[ot * 16 + lm],      asum[ot]);
                atomicAdd(&stats[64 + ot * 16 + lm], asq[ot]);
            }
        }
    } else {
        // max over k=20 per (point, channel); write out[B,O,N]
#pragma unroll
        for (int q = 0; q < 8; ++q) {
            int pidx = l * 8 + q;                 // 0..255 = 4 points x 64 ch
            int nl = pidx >> 6, o = pidx & 63;
            float m = -3.4e38f;
#pragma unroll
            for (int kk = 0; kk < KN; ++kk)
                m = fmaxf(m, hbuf[(nl * KN + kk) * OO + o]);
            out[(size_t)(b * OO + o) * NN + n0 + nl] = m;
        }
    }
}

// ---------------------------------------------------------------------------
// Kernel 5: finalize BN -> folded scale/shift per channel
// ---------------------------------------------------------------------------
__global__ void k_finalize(const float* __restrict__ gamma,
                           const float* __restrict__ beta,
                           float* __restrict__ stats) {
    int o = threadIdx.x;
    if (o < OO) {
        const float cnt  = (float)BB * NN * KN;
        float mean = stats[o] / cnt;
        float var  = stats[64 + o] / cnt - mean * mean;
        float rstd = rsqrtf(var + 1e-5f);
        float sc   = gamma[o] * rstd;
        stats[128 + o] = sc;
        stats[192 + o] = beta[o] - mean * sc;
    }
}

// ---------------------------------------------------------------------------
// Workspace layout (bytes, 256-aligned):
//   xt bf16 : B*N*C*2  = 4,194,304
//   xx f32  : B*N*4    =   131,072
//   idx i32 : B*N*K*4  = 2,621,440
//   Wt bf16 : O*2C*2   =    16,384
//   stats   : 256 f32  =     1,024   (~7 MB total)
// ---------------------------------------------------------------------------
#define OFF_XT 0
#define OFF_XX (OFF_XT + (size_t)BB * NN * CC * 2)
#define OFF_IDX (OFF_XX + (size_t)BB * NN * 4)
#define OFF_WT (OFF_IDX + (size_t)BB * NN * KN * 4)
#define OFF_ST (OFF_WT + (size_t)OO * TWO_C * 2)

extern "C" void kernel_launch(void* const* d_in, const int* in_sizes, int n_in,
                              void* d_out, int out_size, void* d_ws, size_t ws_size,
                              hipStream_t stream) {
    (void)in_sizes; (void)n_in; (void)out_size; (void)ws_size;
    const float* x     = (const float*)d_in[0];
    const float* W     = (const float*)d_in[1];
    const float* gamma = (const float*)d_in[2];
    const float* beta  = (const float*)d_in[3];
    float* out = (float*)d_out;

    char* ws = (char*)d_ws;
    __hip_bfloat16* xt = (__hip_bfloat16*)(ws + OFF_XT);
    float*          xx = (float*)(ws + OFF_XX);
    int*            idx = (int*)(ws + OFF_IDX);
    __hip_bfloat16* Wt = (__hip_bfloat16*)(ws + OFF_WT);
    float*          stats = (float*)(ws + OFF_ST);

    k_transpose<<<BB * NN / 256, 256, 0, stream>>>(x, xt, xx);
    k_prep<<<1, 256, 0, stream>>>(W, Wt, stats);
    k_knn<<<BB * (NN / 32), 32, 0, stream>>>(xt, xx, idx);
    k_conv<1><<<BB * (NN / 4), 32, 0, stream>>>(xt, idx, Wt, stats, nullptr);
    k_finalize<<<1, 64, 0, stream>>>(gamma, beta, stats);
    k_conv<2><<<BB * (NN / 4), 32, 0, stream>>>(xt, idx, Wt, stats, out);
}
